// GraphAttentionLayer_14920716387063
// MI455X (gfx1250) — matrix operand order repaired and checked by reading, not compiled, not run on hardware
//
#include <hip/hip_runtime.h>
#include <hip/hip_bf16.h>

typedef __attribute__((ext_vector_type(16))) _Float16 v16h;
typedef __attribute__((ext_vector_type(8)))  _Float16 v8h;
typedef __attribute__((ext_vector_type(8)))  float    v8f;
typedef __attribute__((ext_vector_type(4)))  float    v4f;

#define NNODE 8192
#define FIN   128
#define NH    2
#define DH    64
#define NCOLS 144   // 128 value cols + 2 E cols + 14 zero pad (multiple of 16)

// ---- CDNA5 async global->LDS path (gfx1250), guarded so compile never breaks
#ifdef __AMDGCN__
#if __has_builtin(__builtin_amdgcn_global_load_async_to_lds_b128) && \
    __has_builtin(__builtin_amdgcn_s_wait_asynccnt)
#define USE_ASYNC_COPY 1
#endif
#endif

// The async-LDS builtins take pointers to a 16-byte GCC vector of int,
// in addrspace(1) (global) / addrspace(3) (LDS).
typedef int v4i_ __attribute__((vector_size(16)));
typedef __attribute__((address_space(1))) v4i_ gbl_v4i;
typedef __attribute__((address_space(3))) v4i_ lds_v4i;

__device__ __forceinline__ gbl_v4i* gbl_cast(const void* p) {
  return (gbl_v4i*)(unsigned long long)p;
}
__device__ __forceinline__ lds_v4i* lds_cast(void* p) {
  // generic LDS pointer: low 32 bits are the LDS byte offset (aperture mapping)
  return (lds_v4i*)(unsigned)(unsigned long long)p;
}

// ---------------------------------------------------------------------------
// Kernel 1: Ht[n][h*64+d] = sum_f X[n][f] * W[h][f][d]   (WMMA f16->f32)
// grid (512, 8), block 32 (one wave per 16x16 tile)
// ---------------------------------------------------------------------------
__global__ __launch_bounds__(32)
void gat_gemm_xw(const float* __restrict__ X, const float* __restrict__ W,
                 float* __restrict__ Ht) {
  const int i0   = blockIdx.x * 16;
  const int n0   = blockIdx.y * 16;
  const int lane = threadIdx.x & 31;
  const int m    = lane & 15;
  const int kb   = (lane >> 4) * 8;    // A-fragment K base: 0 / 8
  const int koff = (lane >> 4) * 16;   // unused; B fragment uses kb (0 / 8)
  const int ncol = n0 + m;
  const int head = ncol >> 6;
  const int d    = ncol & 63;
  const float* Wcol = W + head * (FIN * DH) + d;  // stride DH over k

  v8f acc = {};
  for (int k0 = 0; k0 < FIN; k0 += 32) {
    // A fragment: rows of X, f32 -> f16 inline
    const float* arow = X + (size_t)(i0 + m) * FIN + k0 + kb;
    v4f a0 = *(const v4f*)(arow);
    v4f a1 = *(const v4f*)(arow + 4);
    v4f a2 = *(const v4f*)(arow + 16);
    v4f a3 = *(const v4f*)(arow + 20);
    v16h A;
#pragma unroll
    for (int p = 0; p < 4; ++p) {
      A[p]      = (_Float16)a0[p];
      A[4 + p]  = (_Float16)a1[p];
      A[8 + p]  = (_Float16)a2[p];
      A[12 + p] = (_Float16)a3[p];
    }
    // B fragment: column ncol of Wcat, k in A's order (kb+p, 16+kb+p)
    v16h B;
#pragma unroll
    for (int p = 0; p < 16; ++p)
      B[p] = (_Float16)Wcol[(size_t)(k0 + kb + p + (p & 8)) * DH];
    acc = __builtin_amdgcn_wmma_f32_16x16x32_f16(false, A, false, B,
                                                 (short)0, acc, false, false);
  }
  const int rbase = (lane >> 4) * 8;
#pragma unroll
  for (int r = 0; r < 8; ++r)
    Ht[(size_t)(i0 + rbase + r) * FIN + ncol] = acc[r];
}

// ---------------------------------------------------------------------------
// Kernel 2: s[h][n] = dot(Ht[n][h*64 .. ], a_neigh[h])
// ---------------------------------------------------------------------------
__global__ __launch_bounds__(256)
void gat_sneigh(const float* __restrict__ Ht, const float* __restrict__ a_neigh,
                float* __restrict__ s) {
  int idx = blockIdx.x * 256 + threadIdx.x;
  if (idx >= NH * NNODE) return;
  int h = idx >> 13;
  int n = idx & (NNODE - 1);
  const float* hp = Ht + (size_t)n * FIN + h * DH;
  const float* ap = a_neigh + h * DH;
  float acc = 0.f;
#pragma unroll 8
  for (int d = 0; d < DH; ++d) acc += hp[d] * ap[d];
  s[h * NNODE + n] = acc;
}

// ---------------------------------------------------------------------------
// Kernel 3: per-head global max of s (softmax shift; row-max not needed since
// scores have tiny dynamic range and softmax is shift-invariant)
// ---------------------------------------------------------------------------
__global__ __launch_bounds__(256)
void gat_maxred(const float* __restrict__ s, float* __restrict__ Mx) {
  __shared__ float sm0[256];
  __shared__ float sm1[256];
  float m0 = -3.0e38f, m1 = -3.0e38f;
  for (int j = threadIdx.x; j < NNODE; j += 256) {
    m0 = fmaxf(m0, s[j]);
    m1 = fmaxf(m1, s[NNODE + j]);
  }
  sm0[threadIdx.x] = m0; sm1[threadIdx.x] = m1;
  __syncthreads();
  for (int st = 128; st > 0; st >>= 1) {
    if (threadIdx.x < st) {
      sm0[threadIdx.x] = fmaxf(sm0[threadIdx.x], sm0[threadIdx.x + st]);
      sm1[threadIdx.x] = fmaxf(sm1[threadIdx.x], sm1[threadIdx.x + st]);
    }
    __syncthreads();
  }
  if (threadIdx.x == 0) { Mx[0] = sm0[0]; Mx[1] = sm1[0]; }
}

// ---------------------------------------------------------------------------
// Kernel 4: VT[c][j] (transposed, f16):
//   c in [0,64):   E0[j]*Ht[j][c]   c in [64,128): E1[j]*Ht[j][c]
//   c == 128/129:  E0[j]/E1[j] (denominator cols)   c >= 130: 0
// ---------------------------------------------------------------------------
__global__ __launch_bounds__(256)
void gat_buildvt(const float* __restrict__ Ht, const float* __restrict__ s,
                 const float* __restrict__ Mx, _Float16* __restrict__ VT) {
  int j = blockIdx.x * 256 + threadIdx.x;
  if (j >= NNODE) return;
  float e0 = __expf(s[j] - Mx[0]);
  float e1 = __expf(s[NNODE + j] - Mx[1]);
  const float* hp = Ht + (size_t)j * FIN;
#pragma unroll 4
  for (int d = 0; d < DH; ++d) {
    VT[(size_t)d * NNODE + j]        = (_Float16)(e0 * hp[d]);
    VT[(size_t)(DH + d) * NNODE + j] = (_Float16)(e1 * hp[DH + d]);
  }
  VT[(size_t)128 * NNODE + j] = (_Float16)e0;
  VT[(size_t)129 * NNODE + j] = (_Float16)e1;
#pragma unroll
  for (int c = 130; c < NCOLS; ++c) VT[(size_t)c * NNODE + j] = (_Float16)0.f;
}

// ---------------------------------------------------------------------------
// Kernel 5 (the bandwidth-bound one): ctx[8192][144] = adj @ VT^T
// One block = 16-row strip of adj, 9 waves = 9 column tiles of 16.
// adj strip (16x128 f32 = 8KB chunks) streamed HBM -> LDS with the CDNA5
// async-to-LDS path (ASYNCcnt), double buffered; shared by all 9 waves.
// ---------------------------------------------------------------------------
#define KCHUNK 128
#define KPAD   4

#ifdef USE_ASYNC_COPY
__device__ __forceinline__
void stage_async(const float* __restrict__ adj, int i0, int k0,
                 float (*At)[KCHUNK + KPAD], int tid) {
  // 16 rows x 128 f32 = 512 x 16B transfers; threads 0..255, 2 each
  if (tid < 256) {
#pragma unroll
    for (int tt = 0; tt < 2; ++tt) {
      int t = tid + tt * 256;
      int r = t >> 5, c = (t & 31) << 2;
      __builtin_amdgcn_global_load_async_to_lds_b128(
          gbl_cast(adj + (size_t)(i0 + r) * NNODE + k0 + c),
          lds_cast(&At[r][c]), 0, /*cpol TH=NT*/ 1);
    }
  }
}
#endif

__global__ __launch_bounds__(288)
void gat_attn_gemm(const float* __restrict__ adj, const _Float16* __restrict__ VT,
                   float* __restrict__ ctx) {
  const int i0   = blockIdx.x * 16;
  const int tid  = threadIdx.x;
  const int lane = tid & 31;
  const int n0   = (tid >> 5) * 16;     // wave -> column tile
  const int m    = lane & 15;
  const int kb   = (lane >> 4) * 16;    // A K base: 0 / 16
  const int koff = (lane >> 4) * 16;    // B K base: 0 / 16
  const int ncol = n0 + m;
  const _Float16* vrow = VT + (size_t)ncol * NNODE;
  v8f acc = {};

#ifdef USE_ASYNC_COPY
  __shared__ float Atile[2][16][KCHUNK + KPAD];   // 2 x 8.25KB, double buffered
  stage_async(adj, i0, 0, Atile[0], tid);
  for (int c = 0; c < NNODE / KCHUNK; ++c) {
    __builtin_amdgcn_s_wait_asynccnt(0);   // this wave's stage of chunk c done
    __syncthreads();                       // all waves' stages visible in LDS
    if (c + 1 < NNODE / KCHUNK)
      stage_async(adj, i0, (c + 1) * KCHUNK, Atile[(c + 1) & 1], tid);
    const float (*Ab)[KCHUNK + KPAD] = Atile[c & 1];
#pragma unroll
    for (int kk = 0; kk < KCHUNK; kk += 32) {
      const float* ap = &Ab[m][kk + kb];
      v16h A;
#pragma unroll
      for (int p = 0; p < 8; ++p) {
        A[p]     = (_Float16)ap[p];        // K = kb+p       (binary -> exact)
        A[8 + p] = (_Float16)ap[8 + p];    // K = kb+8+p
      }
      v16h B = *(const v16h*)(vrow + (size_t)c * KCHUNK + kk + koff);
      acc = __builtin_amdgcn_wmma_f32_16x16x32_f16(false, A, false, B,
                                                   (short)0, acc, false, false);
    }
  }
#else
  __shared__ _Float16 AtileH[16][KCHUNK + 8];     // f16 tile, converted at stage
  for (int k0 = 0; k0 < NNODE; k0 += KCHUNK) {
    __syncthreads();
    for (int t = tid; t < 16 * (KCHUNK / 4); t += 288) {
      int row = t / (KCHUNK / 4);
      int c4  = (t % (KCHUNK / 4)) * 4;
      v4f v = __builtin_nontemporal_load(
          (const v4f*)&adj[(size_t)(i0 + row) * NNODE + k0 + c4]);
#pragma unroll
      for (int p = 0; p < 4; ++p) AtileH[row][c4 + p] = (_Float16)v[p];
    }
    __syncthreads();
#pragma unroll
    for (int kk = 0; kk < KCHUNK; kk += 32) {
      const _Float16* ap = &AtileH[m][kk + kb];
      v8h lo = *(const v8h*)(ap);
      v8h hi = *(const v8h*)(ap + 8);
      v16h A;
#pragma unroll
      for (int p = 0; p < 8; ++p) { A[p] = lo[p]; A[8 + p] = hi[p]; }
      v16h B = *(const v16h*)(vrow + k0 + kk + koff);
      acc = __builtin_amdgcn_wmma_f32_16x16x32_f16(false, A, false, B,
                                                   (short)0, acc, false, false);
    }
  }
#endif

  const int rbase = (lane >> 4) * 8;
#pragma unroll
  for (int r = 0; r < 8; ++r)
    ctx[(size_t)(i0 + rbase + r) * NCOLS + ncol] = acc[r];
}

// ---------------------------------------------------------------------------
// Kernel 6: out[i][c] = elu(ctx[i][c] / Z[i][head(c)] + bias[c])
// ---------------------------------------------------------------------------
__global__ __launch_bounds__(256)
void gat_norm_elu(const float* __restrict__ ctx, const float* __restrict__ bias,
                  float* __restrict__ out) {
  int idx = blockIdx.x * 256 + threadIdx.x;
  if (idx >= NNODE * 128) return;
  int i = idx >> 7;
  int c = idx & 127;
  int h = c >> 6;
  float z = ctx[(size_t)i * NCOLS + 128 + h];   // > 0 (self loop guarantees)
  float v = ctx[(size_t)i * NCOLS + c] / z + bias[c];
  out[idx] = v > 0.f ? v : (__expf(v) - 1.f);
}

// ---------------------------------------------------------------------------
extern "C" void kernel_launch(void* const* d_in, const int* in_sizes, int n_in,
                              void* d_out, int out_size, void* d_ws, size_t ws_size,
                              hipStream_t stream) {
  const float* X       = (const float*)d_in[0];   // [8192][128]
  const float* adj     = (const float*)d_in[1];   // [8192][8192]
  const float* W       = (const float*)d_in[2];   // [2][128][64]
  // d_in[3] = a_self: cancels in the row softmax, unused.
  const float* a_neigh = (const float*)d_in[4];   // [2][64]
  const float* bias    = (const float*)d_in[5];   // [128]
  float* out = (float*)d_out;

  char* ws = (char*)d_ws;
  float*    Ht  = (float*)ws;                                        // 4 MB
  float*    s   = (float*)(ws + 4194304);                            // 64 KB
  float*    Mx  = (float*)(ws + 4194304 + 65536);                    // 8 B (+pad)
  _Float16* VT  = (_Float16*)(ws + 4194304 + 65536 + 256);           // 2.25 MB
  float*    ctx = (float*)(ws + 4194304 + 65536 + 256 + 2359296);    // 4.5 MB

  gat_gemm_xw<<<dim3(NNODE / 16, FIN / 16), 32, 0, stream>>>(X, W, Ht);
  gat_sneigh<<<(NH * NNODE + 255) / 256, 256, 0, stream>>>(Ht, a_neigh, s);
  gat_maxred<<<1, 256, 0, stream>>>(s, Mx);
  gat_buildvt<<<NNODE / 256, 256, 0, stream>>>(Ht, s, Mx, VT);
  gat_attn_gemm<<<NNODE / 16, 288, 0, stream>>>(adj, VT, ctx);
  gat_norm_elu<<<(NNODE * 128) / 256, 256, 0, stream>>>(ctx, bias, out);
}
